// HawqTinyClassifier_33036888440887
// MI455X (gfx1250) — 8-bit matrix operand order repaired and checked by reading, compiled, not run on hardware
//
#include <hip/hip_runtime.h>


typedef __attribute__((ext_vector_type(8))) int   v8i;
typedef __attribute__((ext_vector_type(4))) float v4f;

#define BATCH 16384
#define D_IN  2000
#define HID   100
#define OUTN  2
#define KPAD  2048
#define NPAD  112
#define NTILE 7
#define KBLK  32
#define KBH   16

#define I_AMAX_SIG 0
#define I_AMAX_W1  1
#define I_AMAX_W2  2
#define I_AMAX_H   3
#define I_AMAX_Y   4
#define I_PSF      8
#define I_INVPSF   9
#define I_BS1      10
#define I_S8       11
#define I_INVS8    12
#define I_BS2      13
#define I_SY       14
#define I_INVSY    15
#define O_COLMU    320
#define O_RSTD     448
#define O_W2Q      576
#define O_B2Q      800
#define OFF_B1I    4096u
#define OFF_W1P    8192u
#define OFF_BNPART 245760u
#define OFF_H      393216u
#define OFF_Y      7733248u

__device__ __forceinline__ int clampi(int v, int lo, int hi) {
    return v < lo ? lo : (v > hi ? hi : v);
}

__global__ void init_kernel(float* wsf) {
    for (int i = threadIdx.x; i < 1024; i += blockDim.x) wsf[i] = 0.0f;
}

__global__ void absmax_kernel(const float* __restrict__ x, long n, unsigned* out) {
    __shared__ float red[256];
    float m = 0.0f;
    long i = (long)blockIdx.x * blockDim.x + threadIdx.x;
    long stride = (long)gridDim.x * blockDim.x;
    for (; i < n; i += stride) m = fmaxf(m, fabsf(x[i]));
    red[threadIdx.x] = m;
    __syncthreads();
    for (int s = 128; s > 0; s >>= 1) {
        if ((int)threadIdx.x < s) red[threadIdx.x] = fmaxf(red[threadIdx.x], red[threadIdx.x + s]);
        __syncthreads();
    }
    if (threadIdx.x == 0) atomicMax(out, __float_as_uint(red[0]));
}

__global__ void scales1_kernel(const float* __restrict__ b1, float* wsf, int* b1i) {
    int t = threadIdx.x;
    float amax_sig = wsf[I_AMAX_SIG];
    float psf = amax_sig * (1.0f / 15.0f);
    float ws1 = wsf[I_AMAX_W1];
    float bs1 = ws1 * psf;
    if (t == 0) {
        wsf[I_PSF]    = psf;
        wsf[I_INVPSF] = 15.0f / amax_sig;
        wsf[I_BS1]    = bs1;
    }
    if (t < 128) b1i[t] = (t < HID) ? clampi((int)rintf(b1[t] / bs1), -2, 1) : 0;
}

__global__ void pack_w1_kernel(const float* __restrict__ W1, const float* wsf,
                               unsigned* __restrict__ w1p) {
    int d = blockIdx.x * 256 + threadIdx.x;
    if (d >= NTILE * KBLK * 256) return;
    float inv_ws1 = 1.0f / wsf[I_AMAX_W1];
    int t    = d >> 13;
    int rem  = d & 8191;
    int kb   = rem >> 8;
    int rem2 = rem & 255;
    int lane = rem2 >> 3;
    int v    = rem2 & 7;
    int n = t * 16 + (lane & 15);
    unsigned packed = 0;
    #pragma unroll
    for (int j = 0; j < 4; ++j) {
        int kin = ((v >> 1) << 4) + ((lane >> 4) << 3) + ((v & 1) << 2) + j;
        int k = kb * 64 + kin;
        int q = 0;
        if (n < HID && k < D_IN)
            q = clampi((int)rintf(W1[n * D_IN + k] * inv_ws1), -2, 1);
        packed |= (unsigned)(q & 0xff) << (8 * j);
    }
    w1p[d] = packed;
}

__device__ __forceinline__ int quant4(v4f f, float inv_psf) {
    int packed = 0;
    #pragma unroll
    for (int j = 0; j < 4; ++j) {
        int q = clampi((int)rintf(f[j] * inv_psf), -16, 15);
        packed |= (q & 0xff) << (8 * j);
    }
    return packed;
}

__device__ __forceinline__ v8i build_a(const float* __restrict__ ap, int kbase,
                                       float inv_psf) {
    v8i a;
    #pragma unroll
    for (int v = 0; v < 8; ++v) {
        int koff = ((v >> 1) << 4) + ((v & 1) << 2);
        a[v] = quant4(*(const v4f*)(ap + kbase + koff), inv_psf);
    }
    return a;
}

__device__ __forceinline__ v8i build_a_tail(const float* __restrict__ ap,
                                            float inv_psf) {
    v8i a;
    a[0] = quant4(*(const v4f*)(ap + 1984 + 0), inv_psf);
    a[1] = quant4(*(const v4f*)(ap + 1984 + 4), inv_psf);
    #pragma unroll
    for (int v = 2; v < 8; ++v) a[v] = 0;
    return a;
}

__device__ __forceinline__ void load_b7(v8i b[NTILE],
                                        const unsigned char* __restrict__ bp,
                                        int koffB) {
    #pragma unroll
    for (int t = 0; t < NTILE; ++t)
        b[t] = *(const v8i*)(bp + t * 32768 + koffB);
}

__device__ __forceinline__ void wmma7(v8i acc[NTILE], v8i a, const v8i b[NTILE]) {
    #pragma unroll
    for (int t = 0; t < NTILE; ++t)
        acc[t] = __builtin_amdgcn_wmma_i32_16x16x64_iu8(
            true, a, true, b[t], acc[t], false, false);
}

__global__ __launch_bounds__(256) void gemm1_kernel(
    const float* __restrict__ sig, const unsigned char* __restrict__ w1p,
    const int* __restrict__ b1i, const float* __restrict__ wsf,
    float* __restrict__ H, unsigned* hmax) {
    __shared__ int   lacc[4 * NTILE * 256];
    __shared__ float red[256];
    const int lane  = threadIdx.x & 31;
    const int wave  = threadIdx.x >> 5;
    const int mwave = wave & 3;
    const int half  = wave >> 2;
    const int row0  = blockIdx.x * 64 + mwave * 16;
    const float inv_psf = wsf[I_INVPSF];
    const float bs1     = wsf[I_BS1];

    v8i acc[NTILE];
    #pragma unroll
    for (int t = 0; t < NTILE; ++t)
        #pragma unroll
        for (int v = 0; v < 8; ++v) acc[t][v] = 0;

    const int khalf = (lane >> 4) << 3;
    const float* ap = sig + (long)(row0 + (lane & 15)) * D_IN + khalf;
    const unsigned char* bp = w1p + (unsigned)(half * KBH) * 1024u + (lane << 5);

    v8i b0[NTILE], b1v[NTILE];
    load_b7(b0, bp, 0);

    const int kb0base = half * KBH * 64;
    #pragma unroll 1
    for (int i = 0; i < 7; ++i) {
        int kbase = kb0base + i * 128;
        int kpf = kbase + 512;
        __builtin_prefetch(ap + (kpf < D_IN - 16 ? kpf : D_IN - 16), 0, 3);

        v8i a0 = build_a(ap, kbase, inv_psf);
        load_b7(b1v, bp, 1024);
        wmma7(acc, a0, b0);
        v8i a1 = build_a(ap, kbase + 64, inv_psf);
        load_b7(b0, bp, 2048);
        wmma7(acc, a1, b1v);
        bp += 2048;
    }
    {
        v8i a0 = build_a(ap, kb0base + 14 * 64, inv_psf);
        load_b7(b1v, bp, 1024);
        wmma7(acc, a0, b0);
        v8i a1;
        if (half == 0) a1 = build_a(ap, 15 * 64, inv_psf);
        else           a1 = build_a_tail(ap, inv_psf);
        wmma7(acc, a1, b1v);
    }

    const int lbase = mwave * (NTILE * 256);
    if (half == 1) {
        #pragma unroll
        for (int t = 0; t < NTILE; ++t)
            #pragma unroll
            for (int v = 0; v < 8; ++v)
                lacc[lbase + t * 256 + v * 32 + lane] = acc[t][v];
    }
    __syncthreads();

    float lmax = 0.0f;
    if (half == 0) {
        const int m0   = (lane >> 4) << 3;
        const int ncol = lane & 15;
        #pragma unroll
        for (int t = 0; t < NTILE; ++t) {
            int n  = t * 16 + ncol;
            int bv = (n < HID) ? b1i[n] : 0;
            #pragma unroll
            for (int v = 0; v < 8; ++v) {
                int sum = acc[t][v] + lacc[lbase + t * 256 + v * 32 + lane] + bv;
                float h = fmaxf((float)sum * bs1, 0.0f);
                H[(long)(row0 + m0 + v) * NPAD + n] = h;
                lmax = fmaxf(lmax, h);
            }
        }
    }
    red[threadIdx.x] = lmax;
    __syncthreads();
    for (int s = 128; s > 0; s >>= 1) {
        if ((int)threadIdx.x < s) red[threadIdx.x] = fmaxf(red[threadIdx.x], red[threadIdx.x + s]);
        __syncthreads();
    }
    if (threadIdx.x == 0) atomicMax(hmax, __float_as_uint(red[0]));
}

__global__ void scales2_kernel(const float* __restrict__ W2,
                               const float* __restrict__ b2, float* wsf) {
    int t = threadIdx.x;
    float ws2 = wsf[I_AMAX_W2];
    if (t == 0) {
        float s8  = wsf[I_AMAX_H] * (1.0f / 127.0f);
        float bs2 = ws2 * s8;
        wsf[I_S8]    = s8;
        wsf[I_INVS8] = 1.0f / s8;
        wsf[I_BS2]   = bs2;
        wsf[O_B2Q]     = (float)clampi((int)rintf(b2[0] / bs2), -2, 1);
        wsf[O_B2Q + 1] = (float)clampi((int)rintf(b2[1] / bs2), -2, 1);
    }
    if (t < OUTN * HID) wsf[O_W2Q + t] = (float)clampi((int)rintf(W2[t] / ws2), -2, 1);
}

__global__ void bn_stats_kernel(const float* __restrict__ H, const float* wsf,
                                float* __restrict__ part) {
    int c = threadIdx.x;
    int rb = blockIdx.x;
    if (c >= NPAD) return;
    float s8 = wsf[I_S8], inv_s8 = wsf[I_INVS8];
    float s = 0.0f, s2 = 0.0f;
    int r0 = rb * 128;
    for (int i = 0; i < 128; ++i) {
        float h = H[(long)(r0 + i) * NPAD + c];
        int q = clampi((int)rintf(h * inv_s8), -128, 127);
        float xq = (float)q * s8;
        s += xq; s2 += xq * xq;
    }
    part[rb * 224 + c * 2]     = s;
    part[rb * 224 + c * 2 + 1] = s2;
}

__global__ void bn_final_kernel(const float* __restrict__ part, float* wsf) {
    int c = threadIdx.x;
    if (c >= NPAD) return;
    float s = 0.0f, s2 = 0.0f;
    for (int i = 0; i < 128; ++i) {
        s  += part[i * 224 + c * 2];
        s2 += part[i * 224 + c * 2 + 1];
    }
    float mu  = s  * (1.0f / (float)BATCH);
    float var = s2 * (1.0f / (float)BATCH) - mu * mu;
    wsf[O_COLMU + c] = mu;
    wsf[O_RSTD + c]  = rsqrtf(var + 1e-5f);
}

__global__ void lin2_kernel(const float* __restrict__ H, const float* __restrict__ wsf,
                            const float* __restrict__ gamma, const float* __restrict__ beta,
                            float* __restrict__ Y, unsigned* ymax) {
    __shared__ float red[256];
    int r = blockIdx.x * 256 + threadIdx.x;
    float s8 = wsf[I_S8], inv_s8 = wsf[I_INVS8], bs2 = wsf[I_BS2];
    const float* mu   = wsf + O_COLMU;
    const float* rstd = wsf + O_RSTD;
    const float* w2q  = wsf + O_W2Q;
    const float* hrow = H + (long)r * NPAD;
    float a0 = 0.0f, a1 = 0.0f;
    #pragma unroll 4
    for (int c = 0; c < HID; ++c) {
        int q = clampi((int)rintf(hrow[c] * inv_s8), -128, 127);
        float xbn = ((float)q * s8 - mu[c]) * rstd[c] * gamma[c] + beta[c];
        float xi = xbn * inv_s8;
        a0 += xi * w2q[c];
        a1 += xi * w2q[HID + c];
    }
    float y0 = fmaxf((a0 + wsf[O_B2Q])     * bs2, 0.0f);
    float y1 = fmaxf((a1 + wsf[O_B2Q + 1]) * bs2, 0.0f);
    Y[r * 2]     = y0;
    Y[r * 2 + 1] = y1;
    red[threadIdx.x] = fmaxf(y0, y1);
    __syncthreads();
    for (int s = 128; s > 0; s >>= 1) {
        if ((int)threadIdx.x < s) red[threadIdx.x] = fmaxf(red[threadIdx.x], red[threadIdx.x + s]);
        __syncthreads();
    }
    if (threadIdx.x == 0) atomicMax(ymax, __float_as_uint(red[0]));
}

__global__ void scales3_kernel(float* wsf) {
    float m = wsf[I_AMAX_Y];
    wsf[I_SY]    = m * (1.0f / 127.0f);
    wsf[I_INVSY] = 127.0f / m;
}

__global__ void quant_out_kernel(const float* __restrict__ Y, const float* wsf,
                                 float* __restrict__ out) {
    int i = blockIdx.x * 256 + threadIdx.x;
    if (i >= BATCH * OUTN) return;
    float s = wsf[I_SY], inv = wsf[I_INVSY];
    int q = clampi((int)rintf(Y[i] * inv), -128, 127);
    out[i] = (float)q * s;
}

extern "C" void kernel_launch(void* const* d_in, const int* in_sizes, int n_in,
                              void* d_out, int out_size, void* d_ws, size_t ws_size,
                              hipStream_t stream) {
    const float* sig   = (const float*)d_in[0];
    const float* W1    = (const float*)d_in[1];
    const float* b1    = (const float*)d_in[2];
    const float* W2    = (const float*)d_in[3];
    const float* b2    = (const float*)d_in[4];
    const float* gamma = (const float*)d_in[5];
    const float* beta  = (const float*)d_in[6];
    float* out = (float*)d_out;

    float*    wsf  = (float*)d_ws;
    unsigned* wsu  = (unsigned*)d_ws;
    int*      b1i  = (int*)((char*)d_ws + OFF_B1I);
    unsigned* w1p  = (unsigned*)((char*)d_ws + OFF_W1P);
    float*    part = (float*)((char*)d_ws + OFF_BNPART);
    float*    H    = (float*)((char*)d_ws + OFF_H);
    float*    Y    = (float*)((char*)d_ws + OFF_Y);

    (void)in_sizes; (void)n_in; (void)out_size; (void)ws_size;

    init_kernel<<<1, 256, 0, stream>>>(wsf);

    absmax_kernel<<<4096, 256, 0, stream>>>(sig, (long)BATCH * D_IN, wsu + I_AMAX_SIG);
    absmax_kernel<<<256,  256, 0, stream>>>(W1,  (long)HID * D_IN,   wsu + I_AMAX_W1);
    absmax_kernel<<<1,    256, 0, stream>>>(W2,  (long)OUTN * HID,   wsu + I_AMAX_W2);

    scales1_kernel<<<1, 128, 0, stream>>>(b1, wsf, b1i);
    pack_w1_kernel<<<(NTILE * KBLK * 256 + 255) / 256, 256, 0, stream>>>(W1, wsf, w1p);

    gemm1_kernel<<<BATCH / 64, 256, 0, stream>>>(
        sig, (const unsigned char*)w1p, b1i, wsf, H, wsu + I_AMAX_H);

    scales2_kernel<<<1, 256, 0, stream>>>(W2, b2, wsf);
    bn_stats_kernel<<<128, 128, 0, stream>>>(H, wsf, part);
    bn_final_kernel<<<1, 128, 0, stream>>>(part, wsf);

    lin2_kernel<<<BATCH / 256, 256, 0, stream>>>(H, wsf, gamma, beta, Y, wsu + I_AMAX_Y);
    scales3_kernel<<<1, 1, 0, stream>>>(wsf);
    quant_out_kernel<<<(BATCH * OUTN + 255) / 256, 256, 0, stream>>>(Y, wsf, out);
}
